// GraphTemporalRNNConv_80762565034489
// MI455X (gfx1250) — matrix operand order repaired and checked by reading, not compiled, not run on hardware
//
#include <hip/hip_runtime.h>
#include <cstdint>
#include <cstddef>

// Problem constants (fixed by the reference).
#define NNODE 8192
#define NEDGE 200000
#define DDIM  400
#define NREL  16
#define NBLK  100
#define SBLK  4
#define KPAD  416          // 400 padded to a multiple of 32 for wmma K
#define KCH   (KPAD / 32)  // 13 k-chunks
#define GDIM  1200         // 3*D GRU gate width
#define NTW   5            // N tiles (16 cols each) per wave: 25=5*5, 75=15*5

typedef __attribute__((ext_vector_type(16))) __bf16 v16bf;
typedef __attribute__((ext_vector_type(8)))  float  v8f;

// ---- fragment packing (ISA 7.12.2 lane layouts) ---------------------------
// A 16x32 (bf16): lane g*16+lr holds row lr; j=0..7 -> K=g*8+0..7,
//                 j=8..15 -> K=16+g*8+0..7.
__device__ __forceinline__ size_t apack_index(int n, int k) {
    int mt = n >> 4, lr = n & 15;
    int kc = k >> 5, kin = k & 31;
    int g  = (kin >> 3) & 1;
    int j  = (kin & 7) | ((kin >> 4) << 3);
    return ((size_t)mt * KCH + kc) * 512 + (size_t)(g * 16 + lr) * 16 + j;
}
// B 32x16 (bf16): lane g*16+col holds column col; j=0..7 -> K=g*8+0..7,
//                 j=8..15 -> K=16+g*8+0..7.
__device__ __forceinline__ size_t bpack_index(int k, int o) {
    int kc = k >> 5, kin = k & 31;
    int g  = (kin >> 3) & 1;
    int j  = (kin & 7) | ((kin >> 4) << 3);
    int nt = o >> 4, col = o & 15;
    return ((size_t)nt * KCH + kc) * 512 + (size_t)(g * 16 + col) * 16 + j;
}

// ---------------------------------------------------------------- utilities
__global__ void zero_kernel(float* __restrict__ p, int n) {
    int i = blockIdx.x * blockDim.x + threadIdx.x;
    if (i < n) p[i] = 0.0f;
}

// ------------------------------------------------------------- edge norms
__global__ void edge_norm_kernel(const int* __restrict__ ei,
                                 const float* __restrict__ ts,
                                 const float* __restrict__ table,
                                 float* __restrict__ normF,
                                 float* __restrict__ normR) {
    int e = blockIdx.x * blockDim.x + threadIdx.x;
    if (e >= NEDGE) return;
    int s = ei[e];
    int d = ei[NEDGE + e];
    float gf = table[((size_t)d * NNODE + s) * 2 + 0];   // fwd: table[dst,src,0]
    float gr = table[((size_t)s * NNODE + d) * 2 + 1];   // rev: table[src,dst,1]
    float t  = ts[e];
    normF[e] = fminf(1.0f / fmaxf(log1pf(t - gf), 1e-10f), 10.0f);
    normR[e] = fminf(1.0f / fmaxf(log1pf(t - gr), 1e-10f), 10.0f);
}

// ------------------------------------------------- RGCN block-diag messages
// One thread per (edge, 4x4 block); gather + 16 FMA + 4 float atomics.
__global__ void msg_kernel(const float* __restrict__ h,
                           const int* __restrict__ s,
                           const int* __restrict__ d,
                           const int* __restrict__ et,
                           const float* __restrict__ Wr,   // (R,B,S,S), this layer
                           const float* __restrict__ norm,
                           float* __restrict__ agg) {
    int idx = blockIdx.x * blockDim.x + threadIdx.x;
    if (idx >= NEDGE * NBLK) return;
    int e   = idx / NBLK;
    int blk = idx - e * NBLK;
    int sn = s[e], dn = d[e];
    float nrm = norm[e];
    const float* hp = h + (size_t)sn * DDIM + blk * SBLK;
    __builtin_prefetch(hp, 0, 1);                        // global_prefetch_b8
    const float4 m = *(const float4*)hp;
    const float* W = Wr + ((size_t)et[e] * NBLK + blk) * (SBLK * SBLK);
    float4 w0 = *(const float4*)(W + 0);
    float4 w1 = *(const float4*)(W + 4);
    float4 w2 = *(const float4*)(W + 8);
    float4 w3 = *(const float4*)(W + 12);
    float o0 = m.x * w0.x + m.y * w1.x + m.z * w2.x + m.w * w3.x;
    float o1 = m.x * w0.y + m.y * w1.y + m.z * w2.y + m.w * w3.y;
    float o2 = m.x * w0.z + m.y * w1.z + m.z * w2.z + m.w * w3.z;
    float o3 = m.x * w0.w + m.y * w1.w + m.z * w2.w + m.w * w3.w;
    float* dst = agg + (size_t)dn * DDIM + blk * SBLK;
    atomicAdd(dst + 0, o0 * nrm);
    atomicAdd(dst + 1, o1 * nrm);
    atomicAdd(dst + 2, o2 * nrm);
    atomicAdd(dst + 3, o3 * nrm);
}

// ------------------------------------ f32 -> packed-bf16 fragment converters
__global__ void cvt_h_pack(const float* __restrict__ h, __bf16* __restrict__ out) {
    int i = blockIdx.x * blockDim.x + threadIdx.x;
    if (i >= NNODE * KPAD) return;
    int n = i / KPAD, k = i - n * KPAD;
    out[apack_index(n, k)] = (k < DDIM) ? (__bf16)h[(size_t)n * DDIM + k] : (__bf16)0.0f;
}

// dyn_emb is (N,1,D,2); pick direction slice, pad K.
__global__ void cvt_hidden_pack(const float* __restrict__ dyn, __bf16* __restrict__ out, int dir) {
    int i = blockIdx.x * blockDim.x + threadIdx.x;
    if (i >= NNODE * KPAD) return;
    int n = i / KPAD, k = i - n * KPAD;
    out[apack_index(n, k)] =
        (k < DDIM) ? (__bf16)dyn[((size_t)n * DDIM + k) * 2 + dir] : (__bf16)0.0f;
}

// w: (DDIM x ncols) row-major -> packed B fragments, zero-padded K rows.
__global__ void cvt_w_pack(const float* __restrict__ w, __bf16* __restrict__ out, int ncols) {
    int i = blockIdx.x * blockDim.x + threadIdx.x;
    if (i >= KPAD * ncols) return;
    int k = i / ncols, o = i - k * ncols;
    out[bpack_index(k, o)] = (k < DDIM) ? (__bf16)w[(size_t)k * ncols + o] : (__bf16)0.0f;
}

// w: (GDIM x DDIM) row-major; B[k,o] = w[o,k] so GEMM computes x @ w.T
__global__ void cvt_wT_pack(const float* __restrict__ w, __bf16* __restrict__ out) {
    int i = blockIdx.x * blockDim.x + threadIdx.x;
    if (i >= KPAD * GDIM) return;
    int k = i / GDIM, o = i - k * GDIM;
    out[bpack_index(k, o)] = (k < DDIM) ? (__bf16)w[(size_t)o * DDIM + k] : (__bf16)0.0f;
}

// ------------------------------------------------------------- WMMA GEMM
// C[M x Nc] = A_packed * B_packed + addend + bias.
// One wave computes a 16 x (16*NTW) strip. Software-pipelined: the A fragment
// and all NTW B fragments of k-chunk kc+1 are issued before the NTW WMMAs of
// chunk kc, so ~12 b128 loads stay in flight behind 5 multi-cycle XDL ops and
// the per-WMMA s_wait_loadcnt no longer serializes on a just-issued load.
__global__ void wmma_gemm_kernel(const __bf16* __restrict__ Ap,
                                 const __bf16* __restrict__ Bp,
                                 const float* __restrict__ addend,
                                 const float* __restrict__ bias,
                                 float* __restrict__ C, int Nc) {
    int wave = (blockIdx.x * blockDim.x + threadIdx.x) >> 5;
    int lane = threadIdx.x & 31;
    int groupsN = (Nc >> 4) / NTW;
    int mt = wave / groupsN;
    int ng = wave - mt * groupsN;
    if (mt >= (NNODE >> 4)) return;          // wave-uniform: EXEC stays all-1s
    int grp = lane >> 4;
    int lr  = lane & 15;

    const __bf16* aBase = Ap + (size_t)mt * KCH * 512 + (size_t)lane * 16;
    const __bf16* bBase = Bp + (size_t)(ng * NTW) * KCH * 512 + (size_t)lane * 16;

    v8f acc[NTW] = {};

    // prologue: chunk 0 fragments
    v16bf aCur = *(const v16bf*)(aBase);
    v16bf bCur[NTW];
    #pragma unroll
    for (int t = 0; t < NTW; ++t)
        bCur[t] = *(const v16bf*)(bBase + (size_t)t * KCH * 512);

    #pragma unroll
    for (int kc = 0; kc < KCH; ++kc) {
        v16bf aNxt;
        v16bf bNxt[NTW];
        if (kc + 1 < KCH) {                  // issue next chunk's loads first
            aNxt = *(const v16bf*)(aBase + (size_t)(kc + 1) * 512);
            #pragma unroll
            for (int t = 0; t < NTW; ++t)
                bNxt[t] = *(const v16bf*)(bBase + ((size_t)t * KCH + kc + 1) * 512);
        }
        #pragma unroll
        for (int t = 0; t < NTW; ++t) {
            acc[t] = __builtin_amdgcn_wmma_f32_16x16x32_bf16(
                /*neg_a=*/false, aCur, /*neg_b=*/false, bCur[t],
                /*c_mod=*/(short)0, acc[t], /*reuse_a=*/false, /*reuse_b=*/false);
        }
        if (kc + 1 < KCH) {
            aCur = aNxt;
            #pragma unroll
            for (int t = 0; t < NTW; ++t) bCur[t] = bNxt[t];
        }
    }

    // C f32 16x16 layout: vgpr r -> row r (lanes 0-15) / r+8 (lanes 16-31).
    #pragma unroll
    for (int t = 0; t < NTW; ++t) {
        int col = (ng * NTW + t) * 16 + lr;
        #pragma unroll
        for (int r = 0; r < 8; ++r) {
            int row = mt * 16 + r + grp * 8;
            size_t o = (size_t)row * Nc + col;
            float v = acc[t][r];
            if (addend) v += addend[o];
            if (bias)   v += bias[col];
            C[o] = v;
        }
    }
}

// ------------------------------------------------------------- GRU gates
__global__ void gru_gate_kernel(const float* __restrict__ gi,
                                const float* __restrict__ gh,
                                const float* __restrict__ dyn,
                                int dir, float* __restrict__ out) {
    int i = blockIdx.x * blockDim.x + threadIdx.x;
    if (i >= NNODE * DDIM) return;
    int n = i / DDIM, d = i - n * DDIM;
    size_t base = (size_t)n * GDIM;
    float ir = gi[base + d], iz = gi[base + DDIM + d], inn = gi[base + 2 * DDIM + d];
    float hr = gh[base + d], hz = gh[base + DDIM + d], hnn = gh[base + 2 * DDIM + d];
    float h  = dyn[((size_t)n * DDIM + d) * 2 + dir];
    float r = 1.0f / (1.0f + __expf(-(ir + hr)));
    float z = 1.0f / (1.0f + __expf(-(iz + hz)));
    float nv = tanhf(inn + r * hnn);
    out[((size_t)n * DDIM + d) * 2 + dir] = (1.0f - z) * nv + z * h;
}

// ---------------------------------------------------------------- driver
extern "C" void kernel_launch(void* const* d_in, const int* in_sizes, int n_in,
                              void* d_out, int out_size, void* d_ws, size_t ws_size,
                              hipStream_t stream) {
    (void)in_sizes; (void)n_in; (void)out_size; (void)ws_size;
    const int*   edge_index = (const int*)d_in[0];
    const int*   edge_type  = (const int*)d_in[1];
    // d_in[2] = node_id == arange(N): identity, unused.
    const float* timestamps = (const float*)d_in[3];
    const float* table      = (const float*)d_in[4];
    const float* static_emb = (const float*)d_in[5];
    const float* dyn_emb    = (const float*)d_in[6];
    const float* W_rel      = (const float*)d_in[7];
    const float* W_loop     = (const float*)d_in[8];
    const float* bvec       = (const float*)d_in[9];
    const float* w_ih       = (const float*)d_in[10];
    const float* w_hh       = (const float*)d_in[11];
    const float* b_ih       = (const float*)d_in[12];
    const float* b_hh       = (const float*)d_in[13];
    float* out = (float*)d_out;

    char* ws = (char*)d_ws;
    size_t off = 0;
    auto alloc = [&](size_t bytes) -> char* {
        char* p = ws + off;
        off = (off + bytes + 255) & ~(size_t)255;
        return p;
    };
    float*  normF = (float*)alloc((size_t)NEDGE * 4);
    float*  normR = (float*)alloc((size_t)NEDGE * 4);
    float*  h1    = (float*)alloc((size_t)NNODE * DDIM * 4);
    float*  h2    = (float*)alloc((size_t)NNODE * DDIM * 4);
    float*  agg   = (float*)alloc((size_t)NNODE * DDIM * 4);
    __bf16* hbf   = (__bf16*)alloc((size_t)NNODE * KPAD * 2);   // packed A
    __bf16* hpbf  = (__bf16*)alloc((size_t)NNODE * KPAD * 2);   // packed A (hidden)
    __bf16* wl0   = (__bf16*)alloc((size_t)KPAD * DDIM * 2);    // packed B
    __bf16* wl1   = (__bf16*)alloc((size_t)KPAD * DDIM * 2);
    __bf16* wihT  = (__bf16*)alloc((size_t)KPAD * GDIM * 2);
    __bf16* whhT  = (__bf16*)alloc((size_t)KPAD * GDIM * 2);
    float*  gi    = (float*)alloc((size_t)NNODE * GDIM * 4);
    float*  gh    = (float*)alloc((size_t)NNODE * GDIM * 4);

    const int T = 256;
    const int gND  = (NNODE * DDIM + T - 1) / T;
    const int gNK  = (NNODE * KPAD + T - 1) / T;
    const int gMsg = (NEDGE * NBLK + T - 1) / T;
    // waves = (M/16) * (tilesN/NTW); 8 waves per 256-thread block.
    const int gGemmD = (NNODE / 16) * ((DDIM / 16) / NTW) / 8;   // 320
    const int gGemmG = (NNODE / 16) * ((GDIM / 16) / NTW) / 8;   // 960

    // One-time weight conversions + edge norms.
    cvt_w_pack<<<(KPAD * DDIM + T - 1) / T, T, 0, stream>>>(W_loop, wl0, DDIM);
    cvt_w_pack<<<(KPAD * DDIM + T - 1) / T, T, 0, stream>>>(W_loop + (size_t)DDIM * DDIM, wl1, DDIM);
    cvt_wT_pack<<<(KPAD * GDIM + T - 1) / T, T, 0, stream>>>(w_ih, wihT);
    cvt_wT_pack<<<(KPAD * GDIM + T - 1) / T, T, 0, stream>>>(w_hh, whhT);
    edge_norm_kernel<<<(NEDGE + T - 1) / T, T, 0, stream>>>(edge_index, timestamps, table, normF, normR);

    const size_t WrelL1 = (size_t)NREL * NBLK * SBLK * SBLK;

    for (int dir = 0; dir < 2; ++dir) {
        const int* s = (dir == 0) ? edge_index : edge_index + NEDGE;
        const int* d = (dir == 0) ? edge_index + NEDGE : edge_index;
        const float* nrm = (dir == 0) ? normF : normR;

        // RGCN layer 0: agg(msg) + x@Wl0 + b0
        zero_kernel<<<gND, T, 0, stream>>>(agg, NNODE * DDIM);
        msg_kernel<<<gMsg, T, 0, stream>>>(static_emb, s, d, edge_type, W_rel, nrm, agg);
        cvt_h_pack<<<gNK, T, 0, stream>>>(static_emb, hbf);
        wmma_gemm_kernel<<<gGemmD, T, 0, stream>>>(hbf, wl0, agg, bvec, h1, DDIM);

        // RGCN layer 1
        zero_kernel<<<gND, T, 0, stream>>>(agg, NNODE * DDIM);
        msg_kernel<<<gMsg, T, 0, stream>>>(h1, s, d, edge_type, W_rel + WrelL1, nrm, agg);
        cvt_h_pack<<<gNK, T, 0, stream>>>(h1, hbf);
        wmma_gemm_kernel<<<gGemmD, T, 0, stream>>>(hbf, wl1, agg, bvec + DDIM, h2, DDIM);

        // GRU: gi = h2 @ w_ih.T + b_ih ; gh = hidden @ w_hh.T + b_hh
        cvt_h_pack<<<gNK, T, 0, stream>>>(h2, hbf);
        wmma_gemm_kernel<<<gGemmG, T, 0, stream>>>(hbf, wihT, nullptr, b_ih, gi, GDIM);
        cvt_hidden_pack<<<gNK, T, 0, stream>>>(dyn_emb, hpbf, dir);
        wmma_gemm_kernel<<<gGemmG, T, 0, stream>>>(hpbf, whhT, nullptr, b_hh, gh, GDIM);
        gru_gate_kernel<<<gND, T, 0, stream>>>(gi, gh, dyn_emb, dir, out);
    }
}
